// StyledConv_18262200943054
// MI455X (gfx1250) — compile-verified, matrix operand order checked by reading, not run on hardware
//
#include <hip/hip_runtime.h>
#include <hip/hip_bf16.h>
#include <math.h>

// ---- CDNA5 vector types ----
typedef __attribute__((ext_vector_type(16))) __bf16 v16bf;
typedef __attribute__((ext_vector_type(8)))  __bf16 v8bf;
typedef __attribute__((ext_vector_type(8)))  float  v8f;
typedef __attribute__((ext_vector_type(4)))  float  v4f;
typedef __attribute__((ext_vector_type(4)))  unsigned int v4u;
typedef __attribute__((ext_vector_type(8)))  int v8i;
typedef __attribute__((ext_vector_type(4)))  int v4i;

#define BATCH 8
#define CI    512
#define CO    512
#define NSP   4096   // 64*64
#define WDIM  512
#define NTAP  9
#define PW    66     // padded width
#define PSP   (PW*PW) // 4356 padded pixels per channel

#if defined(__HIP_DEVICE_COMPILE__) && defined(__gfx1250__) && \
    __has_builtin(__builtin_amdgcn_tensor_load_to_lds) &&      \
    __has_builtin(__builtin_amdgcn_s_wait_tensorcnt)
#define USE_TDM 1
#else
#define USE_TDM 0
#endif

// ============================================================
// Kernel 1: s'[b][ci] = conv_scale * (eq_scale * <style[b], mod_w[ci]> + mod_bias[ci])
// ============================================================
__global__ void k_modulate(const float* __restrict__ style,
                           const float* __restrict__ mod_weight,
                           const float* __restrict__ mod_bias,
                           float* __restrict__ sprime) {
  int gid = blockIdx.x * blockDim.x + threadIdx.x;   // 0..4095
  int b  = gid >> 9;
  int ci = gid & 511;
  const float eq_scale   = 0.04419417382415922f;     // 1/sqrt(512)
  const float conv_scale = 0.014731391274719742f;    // 1/sqrt(512*9)
  const v4f* st = (const v4f*)(style + (size_t)b * WDIM);
  const v4f* mw = (const v4f*)(mod_weight + (size_t)ci * WDIM);
  float acc = 0.f;
  for (int j = 0; j < WDIM / 4; ++j) {
    v4f a = st[j], c = mw[j];
    acc += a.x * c.x + a.y * c.y + a.z * c.z + a.w * c.w;
  }
  sprime[gid] = conv_scale * (eq_scale * acc + mod_bias[ci]);
}

// ============================================================
// Kernel 2: per-(b,co) demodulation, emit bf16 weights [b][tap][co][ci]
// ============================================================
__global__ __launch_bounds__(256)
void k_demod_weights(const float* __restrict__ weight,   // [CO][CI][3][3]
                     const float* __restrict__ sprime,   // [B][CI]
                     __bf16* __restrict__ wmod) {        // [B][9][CO][CI]
  int blk = blockIdx.x;           // b*512 + co
  int b  = blk >> 9;
  int co = blk & 511;
  int tid = threadIdx.x;
  const float* wco = weight + (size_t)co * (CI * NTAP);  // 4608 floats
  const float* sp  = sprime + (size_t)b * CI;

  float vals[18];
  float ssq = 0.f;
#pragma unroll
  for (int i = 0; i < 18; ++i) {
    int e  = tid + i * 256;       // e = ci*9 + tap
    int ci = e / 9;
    float v = wco[e] * sp[ci];
    vals[i] = v;
    ssq += v * v;
  }

  __shared__ float red[256];
  red[tid] = ssq;
  __syncthreads();
  for (int s = 128; s > 0; s >>= 1) {
    if (tid < s) red[tid] += red[tid + s];
    __syncthreads();
  }
  float demod = rsqrtf(red[0] + 1e-8f);

#pragma unroll
  for (int i = 0; i < 18; ++i) {
    int e   = tid + i * 256;
    int ci  = e / 9;
    int tap = e - ci * 9;
    size_t idx = (((size_t)b * NTAP + tap) * CO + co) * CI + ci;
    wmod[idx] = (__bf16)(vals[i] * demod);
  }
}

// ============================================================
// Kernel 3: zero-padded bf16 copy of x: xpad[b][ci][66][66]
// ============================================================
__global__ void k_pad_x(const float* __restrict__ x,
                        __bf16* __restrict__ xpad, int total) {
  int gid = blockIdx.x * 256 + threadIdx.x;
  if (gid >= total) return;
  int p  = gid % PSP;
  int bc = gid / PSP;           // b*512 + ci
  int hp = p / PW, wp = p % PW;
  float v = 0.f;
  if (hp >= 1 && hp <= 64 && wp >= 1 && wp <= 64)
    v = x[(size_t)bc * NSP + (hp - 1) * 64 + (wp - 1)];
  xpad[gid] = (__bf16)v;
}

// ============================================================
// Kernel 4: implicit-GEMM conv, WMMA bf16 16x16x32, double-buffered LDS,
// TDM (tensor_load_to_lds) for the A tile when available,
// fused noise + bias + leaky_relu * sqrt(2)
// ============================================================
__global__ __launch_bounds__(256)
void k_conv_gemm(const __bf16* __restrict__ xpad,    // [B][CI][66][66]
                 const __bf16* __restrict__ wmod,    // [B][9][CO][CI]
                 const float* __restrict__ noise,    // [B][CO][64][64]
                 const float* __restrict__ noise_weight,
                 const float* __restrict__ act_bias, // [CO]
                 float* __restrict__ out) {          // [B][CO][64][64]
  __shared__ __bf16 lds_a[2][128][32];   // [buf][co_row][k]
  __shared__ __bf16 lds_b[2][128][32];   // [buf][pixel][k]

  const int tid  = threadIdx.x;
  const int nb0  = blockIdx.x * 128;   // pixel tile base
  const int cob  = blockIdx.y * 128;   // CO tile base
  const int b    = blockIdx.z;
  const int lane = tid & 31;
  const int wave = tid >> 5;
  const int rA   = (wave & 1) * 64;    // wave M offset
  const int rB   = (wave >> 1) * 32;   // wave N offset
  const int lrow = lane & 15;
  const int kp   = (lane >> 4) * 8;

  const unsigned short* xpb = (const unsigned short*)(xpad + (size_t)b * CI * PSP);
  const __bf16* wbase = wmod + ((size_t)b * NTAP * CO + cob) * CI;

  // B staging per-thread invariants: pixel fixed across iterations
  const int npix    = tid & 127;
  const int ng      = nb0 + npix;
  const int pixbase = (ng >> 6) * PW + (ng & 63);   // unpadded (h*66+w)
  const int kkbase  = tid >> 7;                     // 0 or 1

  auto stageB = [&](int buf, int tap, int ci0) {
    const int off = (tap / 3) * PW + (tap % 3);     // (dh+1)*66 + (dw+1)
    const unsigned short* src = xpb + (size_t)ci0 * PSP + pixbase + off;
#pragma unroll
    for (int i = 0; i < 8; ++i) {
      int kk = kkbase + 2 * i;                      // pair index 0..15
      unsigned int lo = src[(size_t)(2 * kk) * PSP];
      unsigned int hi = src[(size_t)(2 * kk + 1) * PSP];
      *(unsigned int*)&lds_b[buf][npix][2 * kk] = lo | (hi << 16);
    }
  };

  auto stageA_manual = [&](int buf, int tap, int ci0) {
    const __bf16* ab = wbase + (size_t)tap * CO * CI + ci0;
#pragma unroll
    for (int i = 0; i < 2; ++i) {
      int c = tid + i * 256;
      int row = c >> 2, seg = c & 3;
      *(v8bf*)&lds_a[buf][row][seg * 8] =
          *(const v8bf*)(ab + (size_t)row * CI + seg * 8);
    }
  };

#if USE_TDM
  // Tensor Data Mover: DMA the 128x32 bf16 A tile (row stride CI) into LDS.
  // D# per ISA 8.3/8.4: 2D tensor, data_size=2B, tile 32x128, dim0_stride=512.
  auto stageA_tdm = [&](int buf, int tap, int ci0) {
    if (wave != 0) return;
    const __bf16* ab = wbase + (size_t)tap * CO * CI + ci0;
    unsigned long long ga = (unsigned long long)(uintptr_t)ab;
    unsigned int ldsoff = (unsigned int)(uintptr_t)&lds_a[buf][0][0];
    v4u g0 = { 1u,                                   // count=1, user mode
               ldsoff,                               // lds_addr
               (unsigned int)ga,                     // global_addr[31:0]
               (unsigned int)(ga >> 32) | (2u << 30) // global_addr[56:32] | type=2
             };
    v8i g1 = { (int)(1u << 16),    // data_size = 1 -> 2 bytes; wg_mask = 0
               (int)(512u << 16),  // tensor_dim0[15:0]=512 (bits 63:48)
               (int)(512u << 16),  // tensor_dim0 hi=0 | tensor_dim1[15:0]=512
               (int)(32u << 16),   // tensor_dim1 hi=0 | tile_dim0=32
               128,                // tile_dim1=128, tile_dim2=0
               512,                // tensor_dim0_stride[31:0]=512
               0, 0 };             // stride hi / tensor_dim1_stride = 0
    v4i gz = {0, 0, 0, 0};
#if __clang_major__ >= 23
    v8i gz8 = {0, 0, 0, 0, 0, 0, 0, 0};
    __builtin_amdgcn_tensor_load_to_lds(g0, g1, gz, gz, gz8, 0);
#else
    __builtin_amdgcn_tensor_load_to_lds(g0, g1, gz, gz, 0);
#endif
  };
#endif

  v8f acc[4][2] = {};

  // ---- prologue: stage iteration 0 into buffer 0 ----
#if USE_TDM
  stageA_tdm(0, 0, 0);
#else
  stageA_manual(0, 0, 0);
#endif
  stageB(0, 0, 0);

  // ---- main pipeline: it = tap*16 + ci0/32, 144 iterations ----
  for (int it = 0; it < 144; ++it) {
    const int buf = it & 1;
#if USE_TDM
    if (wave == 0) __builtin_amdgcn_s_wait_tensorcnt(0);
#endif
    __syncthreads();   // staged data for `it` visible; buf^1 free to refill

    if (it + 1 < 144) {
      const int tap1 = (it + 1) >> 4;
      const int ci01 = ((it + 1) & 15) << 5;
#if USE_TDM
      stageA_tdm(buf ^ 1, tap1, ci01);
#else
      stageA_manual(buf ^ 1, tap1, ci01);
#endif
      stageB(buf ^ 1, tap1, ci01);
    }

    // ---- fragments (ISA 16-bit A/B VGPR layout) ----
    v16bf afrag[4], bfrag[2];
#pragma unroll
    for (int mi = 0; mi < 4; ++mi) {
      const __bf16* p = &lds_a[buf][rA + mi * 16 + lrow][kp];
      v8bf lo = *(const v8bf*)p;
      v8bf hi = *(const v8bf*)(p + 16);
#pragma unroll
      for (int j = 0; j < 8; ++j) { afrag[mi][j] = lo[j]; afrag[mi][8 + j] = hi[j]; }
    }
#pragma unroll
    for (int ni = 0; ni < 2; ++ni) {
      const __bf16* p = &lds_b[buf][rB + ni * 16 + lrow][kp];
      v8bf lo = *(const v8bf*)p;
      v8bf hi = *(const v8bf*)(p + 16);
#pragma unroll
      for (int j = 0; j < 8; ++j) { bfrag[ni][j] = lo[j]; bfrag[ni][8 + j] = hi[j]; }
    }

#pragma unroll
    for (int mi = 0; mi < 4; ++mi)
#pragma unroll
      for (int ni = 0; ni < 2; ++ni)
        acc[mi][ni] = __builtin_amdgcn_wmma_f32_16x16x32_bf16(
            false, afrag[mi], false, bfrag[ni],
            (short)0, acc[mi][ni], false, false);
  }

  // ---- fused epilogue: noise + bias + leaky_relu * sqrt(2) ----
  const float nw = noise_weight[0];
  const float SQRT2 = 1.41421356237309515f;
#pragma unroll
  for (int mi = 0; mi < 4; ++mi) {
#pragma unroll
    for (int ni = 0; ni < 2; ++ni) {
#pragma unroll
      for (int r = 0; r < 8; ++r) {
        // C/D layout: VGPR r -> M = r (lanes 0-15) / r+8 (lanes 16-31), N = lane%16
        int m = rA + mi * 16 + r + (lane >> 4) * 8;
        int n = rB + ni * 16 + (lane & 15);
        int co = cob + m;
        size_t idx = ((size_t)b * CO + co) * NSP + (nb0 + n);
        float v = acc[mi][ni][r]
                + nw * __builtin_nontemporal_load(&noise[idx])
                + act_bias[co];
        v = (v > 0.f ? v : 0.2f * v) * SQRT2;
        __builtin_nontemporal_store(v, &out[idx]);
      }
    }
  }
}

// ============================================================
extern "C" void kernel_launch(void* const* d_in, const int* in_sizes, int n_in,
                              void* d_out, int out_size, void* d_ws, size_t ws_size,
                              hipStream_t stream) {
  const float* x        = (const float*)d_in[0];
  const float* style    = (const float*)d_in[1];
  const float* noise    = (const float*)d_in[2];
  const float* weight   = (const float*)d_in[3];
  const float* mod_w    = (const float*)d_in[4];
  const float* mod_b    = (const float*)d_in[5];
  const float* noise_w  = (const float*)d_in[6];
  const float* act_bias = (const float*)d_in[7];
  float* out = (float*)d_out;

  // workspace layout:
  //   [0, 16KB)                 sprime f32 [8][512]
  //   [16KB, 16KB+36MB)         wmod bf16 [8][9][512][512]
  //   [+36MB, +36MB+34MB)       xpad bf16 [8][512][66][66]
  float*  sprime = (float*)d_ws;
  __bf16* wmod   = (__bf16*)((char*)d_ws + 16384);
  __bf16* xpad   = (__bf16*)((char*)d_ws + 16384 + (size_t)BATCH * NTAP * CO * CI * 2);

  k_modulate<<<dim3(16), dim3(256), 0, stream>>>(style, mod_w, mod_b, sprime);
  k_demod_weights<<<dim3(BATCH * CO), dim3(256), 0, stream>>>(weight, sprime, wmod);
  const int pad_total = BATCH * CI * PSP;
  k_pad_x<<<dim3((pad_total + 255) / 256), dim3(256), 0, stream>>>(x, xpad, pad_total);

  dim3 grid(NSP / 128, CO / 128, BATCH);
  k_conv_gemm<<<grid, dim3(256), 0, stream>>>(xpad, wmod, noise, noise_w, act_bias, out);
}
